// QMixtralSparseMoeBlock_29317446762578
// MI455X (gfx1250) — matrix operand order repaired and checked by reading, not compiled, not run on hardware
//
#include <hip/hip_runtime.h>

// ---------------------------------------------------------------------------
// Mixtral sparse MoE block for MI455X (gfx1250, wave32, WMMA bf16 path)
//   - routed top-2 experts, compacted token lists
//   - v_wmma_f32_16x16x32_bf16 matrix core path, f32 accumulation
//   - GLOBAL_LOAD_ASYNC_TO_LDS_B128 for gathered activation tiles (ASYNCcnt)
//   - TENSOR_LOAD_TO_LDS (TDM) for contiguous h tiles with LDS padding (TENSORcnt)
// ---------------------------------------------------------------------------

typedef __attribute__((ext_vector_type(8)))  float        v8f;
typedef __attribute__((ext_vector_type(16))) __bf16       v16bf;
typedef __attribute__((ext_vector_type(4)))  unsigned int u32x4;
typedef __attribute__((ext_vector_type(8)))  int          i32x8;
typedef __attribute__((ext_vector_type(4)))  int          i32x4;

enum : int {
  T_TOK = 8192,   // 4 * 2048 tokens
  H_DIM = 1024,
  F_DIM = 4096,
  NEXP  = 8,
  BM    = 128,    // block tile M
  BN    = 64,     // block tile N
  BK    = 32,     // K step (one wmma_f32_16x16x32_bf16 K)
  LDE   = 40      // LDS row stride in bf16 elems (32 + 8 pad, keeps 16B align)
};

static __device__ __forceinline__ int imin(int a, int b) { return a < b ? a : b; }

// float -> bf16 round-to-nearest-even
static __device__ __forceinline__ unsigned short f2bf(float x) {
  unsigned u = __float_as_uint(x);
  unsigned r = ((u >> 16) & 1u) + 0x7FFFu;
  return (unsigned short)((u + r) >> 16);
}

// Async global->LDS 16B copy (per-lane addresses, tracked by ASYNCcnt)
static __device__ __forceinline__ void async_copy_b128(unsigned lds_off, const void* gptr) {
  asm volatile("global_load_async_to_lds_b128 %0, %1, off"
               :: "v"(lds_off), "v"(gptr) : "memory");
}
static __device__ __forceinline__ void wait_async0() {
  asm volatile("s_wait_asynccnt 0x0" ::: "memory");
}

// A operand (16x32 bf16, dense): lane L holds row M = L&15.
// lane<16: K in {0..7, 16..23}; lane>=16: K in {8..15, 24..31}  (ISA 7.12.2)
static __device__ __forceinline__ v16bf load_frag_a(const unsigned short* base, int lane) {
  int r   = lane & 15;
  int sel = (lane >> 4) & 1;
  const unsigned short* p = base + r * LDE + sel * 8;
  union { v16bf v; uint4 q[2]; } f;
  f.q[0] = *(const uint4*)(p);        // K sel*8   .. sel*8+7
  f.q[1] = *(const uint4*)(p + 16);   // K sel*8+16.. sel*8+23
  return f.v;
}

// B operand (32x16 bf16, dense): lane L holds column N = L&15.
// lane<16: K in {0..7, 16..23}; lane>=16: K in {8..15, 24..31} (matches A)
static __device__ __forceinline__ v16bf load_frag_b(const unsigned short* base, int lane) {
  int n    = lane & 15;
  int half = (lane >> 4) & 1;
  const unsigned short* p = base + n * LDE + half * 8;
  union { v16bf v; uint4 q[2]; } f;
  f.q[0] = *(const uint4*)(p);
  f.q[1] = *(const uint4*)(p + 16);
  return f.v;
}

static __device__ __forceinline__ v8f zero_v8f() {
  v8f z;
#pragma unroll
  for (int i = 0; i < 8; ++i) z[i] = 0.0f;
  return z;
}

// ---------------------------------------------------------------------------
// Router: logits = x @ gate_w.T ; softmax top-2 ; combine weights ; token lists
// ---------------------------------------------------------------------------
__global__ __launch_bounds__(64) void moe_router(
    const float* __restrict__ x, const float* __restrict__ gw,
    float* __restrict__ logits_out, float* __restrict__ combine,
    int* __restrict__ counts, int* __restrict__ toklist) {
  __shared__ float red[NEXP][64];
  __shared__ float lg[NEXP];
  const int t = blockIdx.x;
  const int tid = threadIdx.x;

  float acc[NEXP];
#pragma unroll
  for (int e = 0; e < NEXP; ++e) acc[e] = 0.0f;

  const float* xr = x + (size_t)t * H_DIM;
  for (int j = tid; j < H_DIM; j += 64) {
    float xv = xr[j];
#pragma unroll
    for (int e = 0; e < NEXP; ++e) acc[e] += xv * gw[e * H_DIM + j];
  }
#pragma unroll
  for (int e = 0; e < NEXP; ++e) red[e][tid] = acc[e];
  __syncthreads();

  if (tid < NEXP) {
    float s = 0.0f;
    for (int i = 0; i < 64; ++i) s += red[tid][i];   // fixed order -> deterministic
    lg[tid] = s;
    logits_out[(size_t)t * NEXP + tid] = s;
  }
  __syncthreads();

  if (tid == 0) {
    int i1 = 0; float l1 = lg[0];
    for (int e = 1; e < NEXP; ++e) if (lg[e] > l1) { l1 = lg[e]; i1 = e; }
    int i2 = (i1 == 0) ? 1 : 0; float l2 = lg[i2];
    for (int e = 0; e < NEXP; ++e)
      if (e != i1 && lg[e] > l2) { l2 = lg[e]; i2 = e; }
    // renormalized top-2 softmax weights (stable: l1 >= l2)
    float e2 = __expf(l2 - l1);
    float w1n = 1.0f / (1.0f + e2);
    float w2n = e2 * w1n;
    float* c = combine + (size_t)t * NEXP;
#pragma unroll
    for (int e = 0; e < NEXP; ++e) c[e] = 0.0f;
    c[i1] = w1n;
    c[i2] = w2n;
    int p1 = atomicAdd(&counts[i1], 1);
    toklist[i1 * T_TOK + p1] = t;
    int p2 = atomicAdd(&counts[i2], 1);
    toklist[i2 * T_TOK + p2] = t;
  }
}

// ---------------------------------------------------------------------------
// Utility kernels
// ---------------------------------------------------------------------------
__global__ void moe_zero_counts(int* c) {
  if (threadIdx.x < NEXP) c[threadIdx.x] = 0;
}

__global__ void moe_cast_x(const float* __restrict__ x, unsigned short* __restrict__ xb, int n4) {
  int i = blockIdx.x * blockDim.x + threadIdx.x;
  if (i < n4) {
    float4 v = ((const float4*)x)[i];
    ushort4 o;
    o.x = f2bf(v.x); o.y = f2bf(v.y); o.z = f2bf(v.z); o.w = f2bf(v.w);
    ((ushort4*)xb)[i] = o;
  }
}

__global__ void moe_zero_out(float* __restrict__ p, int n4) {
  int i = blockIdx.x * blockDim.x + threadIdx.x;
  if (i < n4) ((float4*)p)[i] = make_float4(0.f, 0.f, 0.f, 0.f);
}

// ---------------------------------------------------------------------------
// GEMM1: h[r,:] = silu(x_r . w1^T) * (x_r . w3^T)   over expert token list
// A: xb (bf16, gathered rows, ASYNC global->LDS)  B: w1/w3 (fp32 -> bf16)
// ---------------------------------------------------------------------------
__global__ __launch_bounds__(256) void moe_gemm1(
    const unsigned short* __restrict__ xb,
    const float* __restrict__ w1e, const float* __restrict__ w3e,
    const int* __restrict__ counts, const int* __restrict__ toklist,
    unsigned short* __restrict__ hbuf, int e) {
  const int cnt = counts[e];
  const int m0 = blockIdx.y * BM;
  if (m0 >= cnt) return;
  const int n0 = blockIdx.x * BN;

  __shared__ __align__(16) unsigned short As[BM * LDE];
  __shared__ __align__(16) unsigned short B1s[BN * LDE];
  __shared__ __align__(16) unsigned short B3s[BN * LDE];
  __shared__ int toks[BM];

  const int tid = threadIdx.x;
  if (tid < BM) {
    int idx = m0 + tid;
    toks[tid] = toklist[e * T_TOK + imin(idx, cnt - 1)];
  }
  __syncthreads();

  // A staging: each thread owns half a row (16 bf16 = 2 x 16B async copies)
  const int arow  = tid >> 1;
  const int ahalf = tid & 1;
  const size_t abase = (size_t)toks[arow] * H_DIM + ahalf * 16;
  const unsigned alds = (unsigned)(size_t)(void*)(As + arow * LDE + ahalf * 16);
  // B staging: each thread owns a quarter row (8 fp32 -> 8 bf16)
  const int brow = tid >> 2;
  const int bq   = tid & 3;
  const size_t bbase = (size_t)(n0 + brow) * H_DIM + bq * 8;

  const int wv   = tid >> 5;          // wave 0..7
  const int lane = tid & 31;
  const int wm   = (wv & 3) * 32;     // wave M origin within block
  const int wn   = (wv >> 2) * 32;    // wave N origin within block

  v8f acc1[2][2], acc3[2][2];
#pragma unroll
  for (int i = 0; i < 2; ++i)
#pragma unroll
    for (int j = 0; j < 2; ++j) { acc1[i][j] = zero_v8f(); acc3[i][j] = zero_v8f(); }

  for (int kk = 0; kk < H_DIM; kk += BK) {
    { // A tile: async gather straight into LDS (no VGPR round trip)
      const unsigned short* gp = xb + abase + kk;
      async_copy_b128(alds,      gp);
      async_copy_b128(alds + 16, gp + 8);
    }
    { // B tiles (fp32 -> bf16 in flight)
      const float4* s1 = (const float4*)(w1e + bbase + kk);
      const float4* s3 = (const float4*)(w3e + bbase + kk);
      float4 u0 = s1[0], u1 = s1[1], t0 = s3[0], t1 = s3[1];
      ushort4 p0, p1, q0, q1;
      p0.x = f2bf(u0.x); p0.y = f2bf(u0.y); p0.z = f2bf(u0.z); p0.w = f2bf(u0.w);
      p1.x = f2bf(u1.x); p1.y = f2bf(u1.y); p1.z = f2bf(u1.z); p1.w = f2bf(u1.w);
      q0.x = f2bf(t0.x); q0.y = f2bf(t0.y); q0.z = f2bf(t0.z); q0.w = f2bf(t0.w);
      q1.x = f2bf(t1.x); q1.y = f2bf(t1.y); q1.z = f2bf(t1.z); q1.w = f2bf(t1.w);
      *(ushort4*)(B1s + brow * LDE + bq * 8)     = p0;
      *(ushort4*)(B1s + brow * LDE + bq * 8 + 4) = p1;
      *(ushort4*)(B3s + brow * LDE + bq * 8)     = q0;
      *(ushort4*)(B3s + brow * LDE + bq * 8 + 4) = q1;
      if (kk + BK < H_DIM) {   // stream-ahead hint -> global_prefetch_b8
        __builtin_prefetch(w1e + bbase + kk + BK, 0, 1);
        __builtin_prefetch(w3e + bbase + kk + BK, 0, 1);
      }
    }
    wait_async0();             // A tile landed in LDS
    __syncthreads();

    v16bf a[2], b1f[2], b3f[2];
    a[0]   = load_frag_a(As  + (wm)      * LDE, lane);
    a[1]   = load_frag_a(As  + (wm + 16) * LDE, lane);
    b1f[0] = load_frag_b(B1s + (wn)      * LDE, lane);
    b1f[1] = load_frag_b(B1s + (wn + 16) * LDE, lane);
    b3f[0] = load_frag_b(B3s + (wn)      * LDE, lane);
    b3f[1] = load_frag_b(B3s + (wn + 16) * LDE, lane);
#pragma unroll
    for (int i = 0; i < 2; ++i)
#pragma unroll
      for (int j = 0; j < 2; ++j) {
        acc1[i][j] = __builtin_amdgcn_wmma_f32_16x16x32_bf16(
            false, a[i], false, b1f[j], (short)0, acc1[i][j], false, false);
        acc3[i][j] = __builtin_amdgcn_wmma_f32_16x16x32_bf16(
            false, a[i], false, b3f[j], (short)0, acc3[i][j], false, false);
      }
    __syncthreads();
  }

  // Epilogue: silu(g1) * g3 -> bf16 hbuf (list-local rows)
#pragma unroll
  for (int i = 0; i < 2; ++i)
#pragma unroll
    for (int j = 0; j < 2; ++j)
#pragma unroll
      for (int r = 0; r < 8; ++r) {
        int ml = wm + i * 16 + ((lane >> 4) << 3) + r;
        int gm = m0 + ml;
        if (gm < cnt) {
          int col = n0 + wn + j * 16 + (lane & 15);
          float g = acc1[i][j][r];
          float s = g * (1.0f / (1.0f + __expf(-g)));  // silu
          float h = s * acc3[i][j][r];
          hbuf[(size_t)gm * F_DIM + col] = f2bf(h);
        }
      }
}

// ---------------------------------------------------------------------------
// GEMM2: out[tok,:] += combine[tok,e] * (h_r . w2^T)
// A: hbuf (bf16, contiguous 128x32 tile -> Tensor Data Mover into padded LDS)
// ---------------------------------------------------------------------------
__global__ __launch_bounds__(256) void moe_gemm2(
    const unsigned short* __restrict__ hbuf,
    const float* __restrict__ w2e,
    const int* __restrict__ counts, const int* __restrict__ toklist,
    const float* __restrict__ combine,
    float* __restrict__ out, int e) {
  const int cnt = counts[e];
  const int m0 = blockIdx.y * BM;
  if (m0 >= cnt) return;
  const int n0 = blockIdx.x * BN;    // within H

  __shared__ __align__(16) unsigned short As[BM * LDE];
  __shared__ __align__(16) unsigned short Bs[BN * LDE];
  __shared__ int toks[BM];

  const int tid = threadIdx.x;
  if (tid < BM) {
    int idx = m0 + tid;
    toks[tid] = toklist[e * T_TOK + imin(idx, cnt - 1)];
  }

  const int brow = tid >> 2;
  const int bq   = tid & 3;
  const size_t bbase = (size_t)(n0 + brow) * F_DIM + bq * 8;

  const int wv   = tid >> 5;
  const int lane = tid & 31;
  const int wm   = (wv & 3) * 32;
  const int wn   = (wv >> 2) * 32;

  // TDM descriptor invariants for the 128x32 bf16 A tile:
  //  data_size=2B, tile 32 x 128, tensor_dim0=32 (exact rows), tensor_dim1=cnt-m0
  //  (rows past the routed count read as zero), row stride = F_DIM elems,
  //  LDS padding: 4 DWORDs every 16 DWORDs -> row stride LDE=40 elems.
  const unsigned alds = (unsigned)(size_t)(void*)As;
  const unsigned long long gabase =
      (unsigned long long)(size_t)(hbuf + (size_t)m0 * F_DIM);
  const int td1 = cnt - m0;

  v8f acc[2][2];
#pragma unroll
  for (int i = 0; i < 2; ++i)
#pragma unroll
    for (int j = 0; j < 2; ++j) acc[i][j] = zero_v8f();

  __syncthreads();

  for (int kk = 0; kk < F_DIM; kk += BK) {
    if (tid < 32) { // wave 0 drives the Tensor Data Mover (EXEC ignored, wave-level op)
      unsigned long long ga = gabase + (unsigned long long)kk * 2ull;
      u32x4 g0;
      g0[0] = 1u;                                        // count=1, no gather
      g0[1] = alds;                                      // lds_addr
      g0[2] = (unsigned)(ga & 0xFFFFFFFFull);            // global_addr[31:0]
      g0[3] = (unsigned)((ga >> 32) & 0x1FFFFFFull)      // global_addr[56:32]
              | (2u << 30);                              // type=2 (image)
      i32x8 g1;
      g1[0] = (int)((1u << 16)      // data_size = 2 bytes
                  | (1u << 20)      // pad_enable
                  | (3u << 22)      // pad_interval: every 16 DWORDs (64B row)
                  | (3u << 25));    // pad_amount: 4 DWORDs (16B)
      g1[1] = (int)(32u << 16);                          // tensor_dim0 = 32
      g1[2] = (int)(((unsigned)td1 & 0xFFFFu) << 16);    // tensor_dim1 lo16
      g1[3] = (int)((((unsigned)td1 >> 16) & 0xFFFFu)    // tensor_dim1 hi16
                  | (32u << 16));                        // tile_dim0 = 32
      g1[4] = 128;                                       // tile_dim1=128, tile_dim2=0
      g1[5] = F_DIM;                                     // tensor_dim0_stride lo32
      g1[6] = 0;
      g1[7] = 0;
      i32x4 gz = {0, 0, 0, 0};
#if defined(__clang_major__) && (__clang_major__ >= 23)
      i32x8 gz8 = {0, 0, 0, 0, 0, 0, 0, 0};
      __builtin_amdgcn_tensor_load_to_lds(g0, g1, gz, gz, gz8, 0);
#else
      __builtin_amdgcn_tensor_load_to_lds(g0, g1, gz, gz, 0);
#endif
    }
    { // B tile (w2 fp32 -> bf16)
      const float4* s2 = (const float4*)(w2e + bbase + kk);
      float4 u0 = s2[0], u1 = s2[1];
      ushort4 p0, p1;
      p0.x = f2bf(u0.x); p0.y = f2bf(u0.y); p0.z = f2bf(u0.z); p0.w = f2bf(u0.w);
      p1.x = f2bf(u1.x); p1.y = f2bf(u1.y); p1.z = f2bf(u1.z); p1.w = f2bf(u1.w);
      *(ushort4*)(Bs + brow * LDE + bq * 8)     = p0;
      *(ushort4*)(Bs + brow * LDE + bq * 8 + 4) = p1;
      if (kk + BK < F_DIM)
        __builtin_prefetch(w2e + bbase + kk + BK, 0, 1);
    }
    __builtin_amdgcn_s_wait_tensorcnt(0);   // TDM tile landed (wave 0; NOP elsewhere)
    __syncthreads();

    v16bf a[2], bf[2];
    a[0]  = load_frag_a(As + (wm)      * LDE, lane);
    a[1]  = load_frag_a(As + (wm + 16) * LDE, lane);
    bf[0] = load_frag_b(Bs + (wn)      * LDE, lane);
    bf[1] = load_frag_b(Bs + (wn + 16) * LDE, lane);
#pragma unroll
    for (int i = 0; i < 2; ++i)
#pragma unroll
      for (int j = 0; j < 2; ++j)
        acc[i][j] = __builtin_amdgcn_wmma_f32_16x16x32_bf16(
            false, a[i], false, bf[j], (short)0, acc[i][j], false, false);
    __syncthreads();
  }

  // Epilogue: scatter-accumulate into out (unique (token,col) per expert kernel;
  // expert kernels are serialized on the stream -> race-free, deterministic)
#pragma unroll
  for (int i = 0; i < 2; ++i)
#pragma unroll
    for (int j = 0; j < 2; ++j)
#pragma unroll
      for (int r = 0; r < 8; ++r) {
        int ml = wm + i * 16 + ((lane >> 4) << 3) + r;
        int gm = m0 + ml;
        if (gm < cnt) {
          int t = toks[ml];
          int col = n0 + wn + j * 16 + (lane & 15);
          float w = combine[(size_t)t * NEXP + e];
          float* o = out + (size_t)t * H_DIM + col;
          *o = *o + w * acc[i][j][r];
        }
      }
}

// ---------------------------------------------------------------------------
// Launcher
// ---------------------------------------------------------------------------
extern "C" void kernel_launch(void* const* d_in, const int* in_sizes, int n_in,
                              void* d_out, int out_size, void* d_ws, size_t ws_size,
                              hipStream_t stream) {
  (void)in_sizes; (void)n_in; (void)out_size; (void)ws_size;
  const float* x  = (const float*)d_in[0];   // [4,2048,1024]
  const float* gw = (const float*)d_in[1];   // [8,1024]
  const float* w1 = (const float*)d_in[2];   // [8,4096,1024]
  const float* w2 = (const float*)d_in[3];   // [8,1024,4096]
  const float* w3 = (const float*)d_in[4];   // [8,4096,1024]

  float* out    = (float*)d_out;                       // [T,H]
  float* logits = out + (size_t)T_TOK * H_DIM;         // [T,E] appended

  // workspace layout
  char* ws = (char*)d_ws;
  size_t off = 0;
  unsigned short* xb = (unsigned short*)(ws + off);    // x in bf16  [T,H]
  off += (size_t)T_TOK * H_DIM * 2;
  unsigned short* hbuf = (unsigned short*)(ws + off);  // h in bf16  [T,F] (per-expert reuse)
  off += (size_t)T_TOK * F_DIM * 2;
  float* combine = (float*)(ws + off);                 // [T,E]
  off += (size_t)T_TOK * NEXP * 4;
  int* toklist = (int*)(ws + off);                     // [E,T]
  off += (size_t)NEXP * T_TOK * 4;
  int* counts = (int*)(ws + off);                      // [E]

  moe_zero_counts<<<1, 32, 0, stream>>>(counts);
  moe_router<<<T_TOK, 64, 0, stream>>>(x, gw, logits, combine, counts, toklist);

  const int n4 = T_TOK * H_DIM / 4;
  moe_cast_x<<<(n4 + 255) / 256, 256, 0, stream>>>(x, xb, n4);
  moe_zero_out<<<(n4 + 255) / 256, 256, 0, stream>>>(out, n4);

  dim3 g1(F_DIM / BN, T_TOK / BM);   // (64, 64)
  dim3 g2(H_DIM / BN, T_TOK / BM);   // (16, 64)
  for (int e = 0; e < NEXP; ++e) {
    const float* w1e = w1 + (size_t)e * F_DIM * H_DIM;
    const float* w3e = w3 + (size_t)e * F_DIM * H_DIM;
    const float* w2e = w2 + (size_t)e * H_DIM * F_DIM;
    moe_gemm1<<<g1, 256, 0, stream>>>(xb, w1e, w3e, counts, toklist, hbuf, e);
    moe_gemm2<<<g2, 256, 0, stream>>>(hbuf, w2e, counts, toklist, combine, out, e);
  }
}
